// CUAIModel_16466904612935
// MI455X (gfx1250) — compile-verified, matrix operand order checked by reading, not run on hardware
//
#include <hip/hip_runtime.h>

#define NODES 100000
#define EDGES 1600000
#define ETOT  (EDGES + NODES)
#define HEADS 8
#define DIM   16
#define HD    128
#define SLOPE 0.2f
#define EPSD  1e-16f

typedef __attribute__((ext_vector_type(16))) __bf16          v16bf;
typedef __attribute__((ext_vector_type(16))) unsigned short  v16us;
typedef __attribute__((ext_vector_type(8)))  float           v8f;

__device__ __forceinline__ unsigned short f2bf(float f) {
  unsigned int u = __float_as_uint(f);
  u = (u + 0x7FFFu + ((u >> 16) & 1u)) >> 16;   // round-to-nearest-even
  return (unsigned short)u;
}

// ------------- pack weights [K,NC] f32 -> WMMA B-fragment order (bf16) -------------
// Layout: Bf[(((nt*NK)+ks)*32 + lane)*16 + e], so a lane's fragment is 32 contiguous B.
template<int K, int NTN>
__global__ void k_pack_b(const float* __restrict__ W, unsigned short* __restrict__ Bf) {
  constexpr int NK = (K + 31) / 32;
  constexpr int NC = NTN * 16;
  int id = blockIdx.x * blockDim.x + threadIdx.x;
  if (id >= NTN * NK * 512) return;
  int e    = id & 15;
  int lane = (id >> 4) & 31;
  int ks   = (id >> 9) % NK;
  int nt   = id / (NK * 512);
  int lo = lane & 15, hi = lane >> 4;
  int i = e >> 1, j = e & 1;
  int k = (i < 4) ? (2 * i + j + 8 * hi) : (16 + 2 * (i - 4) + j + 8 * hi);
  int kg = ks * 32 + k;
  unsigned short v = 0;
  if (kg < K) v = f2bf(W[kg * NC + nt * 16 + lo]);
  Bf[id] = v;
}

// ------------- WMMA GEMM: one wave computes a 16 x (NTN*16) output strip -------------
// A fragment per lane = two contiguous 8-float runs of its A row -> four b128 loads,
// converted with native bf16 casts (v_cvt_*bf16* on gfx1250).
template<int K, int NTN>
__global__ void k_gemm(const float* __restrict__ A, const unsigned short* __restrict__ Bf,
                       float* __restrict__ C, const float* __restrict__ bias, int totalWaves) {
  constexpr int NK = (K + 31) / 32;
  constexpr int NC = NTN * 16;
  int wave = blockIdx.x * (blockDim.x >> 5) + (threadIdx.x >> 5);
  if (wave >= totalWaves) return;                // wave-uniform guard
  int lane = threadIdx.x & 31;
  int lo = lane & 15;
  int hi = lane >> 4;
  int mt = wave;                                 // one wave per row tile
  int row = mt * 16 + lo;

  v8f acc[NTN];
#pragma unroll
  for (int nt = 0; nt < NTN; ++nt) acc[nt] = (v8f){};

#pragma unroll
  for (int ks = 0; ks < NK; ++ks) {
    int kk = ks * 32;
    const float* ar = A + row * K + kk + 8 * hi;
    float4 p0 = *(const float4*)(ar);
    float4 p1 = *(const float4*)(ar + 4);
    float4 p2, p3;
    if constexpr (K >= 32) {                     // upper 16-K chunk exists
      p2 = *(const float4*)(ar + 16);
      p3 = *(const float4*)(ar + 20);
    } else {                                     // K=16: zero-padded to K=32
      p2 = make_float4(0.f, 0.f, 0.f, 0.f);
      p3 = p2;
    }
    v16bf ab;
    ab[0]  = (__bf16)p0.x; ab[1]  = (__bf16)p0.y; ab[2]  = (__bf16)p0.z; ab[3]  = (__bf16)p0.w;
    ab[4]  = (__bf16)p1.x; ab[5]  = (__bf16)p1.y; ab[6]  = (__bf16)p1.z; ab[7]  = (__bf16)p1.w;
    ab[8]  = (__bf16)p2.x; ab[9]  = (__bf16)p2.y; ab[10] = (__bf16)p2.z; ab[11] = (__bf16)p2.w;
    ab[12] = (__bf16)p3.x; ab[13] = (__bf16)p3.y; ab[14] = (__bf16)p3.z; ab[15] = (__bf16)p3.w;
#pragma unroll
    for (int nt = 0; nt < NTN; ++nt) {           // reuse A across 8 column tiles
      union { v16us u; v16bf b; } cb;
      cb.u = *(const v16us*)(Bf + (((nt * NK) + ks) * 32 + lane) * 16);
      acc[nt] = __builtin_amdgcn_wmma_f32_16x16x32_bf16(false, ab, false, cb.b,
                                                        (short)0, acc[nt], false, false);
    }
  }
#pragma unroll
  for (int nt = 0; nt < NTN; ++nt) {
    int col = nt * 16 + lo;
    float bv = bias ? bias[col] : 0.f;
#pragma unroll
    for (int r = 0; r < 8; ++r)
      C[(mt * 16 + r + 8 * hi) * NC + col] = acc[nt][r] + bv;
  }
}

// ---------------- per-node attention coefficients + state init ----------------
__global__ void k_att_prep(const float* __restrict__ g, const float* __restrict__ attS,
                           const float* __restrict__ attD, const float* __restrict__ bias,
                           float* __restrict__ aS, float* __restrict__ aD,
                           float* __restrict__ mM, float* __restrict__ den,
                           float* __restrict__ acc) {
  int id = blockIdx.x * blockDim.x + threadIdx.x;      // N*H threads
  if (id >= NODES * HEADS) return;
  int n = id >> 3, h = id & 7;
  const float* gp = g + n * HD + h * DIM;
  const float* sv = attS + h * DIM;
  const float* dv = attD + h * DIM;
  float s = 0.f, d = 0.f;
#pragma unroll
  for (int k = 0; k < DIM; ++k) { s += gp[k] * sv[k]; d += gp[k] * dv[k]; }
  aS[id] = s; aD[id] = d;
  mM[id] = -__builtin_inff();
  den[id] = 0.f;
  float* ap = acc + n * HD + h * DIM;
  const float* bp = bias + h * DIM;
#pragma unroll
  for (int k = 0; k < DIM; ++k) ap[k] = bp[k];         // out starts at +bias
}

__device__ __forceinline__ void edge_nodes(const long long* __restrict__ ei, int e, int& s, int& t) {
  if (e < EDGES) { s = (int)ei[e]; t = (int)ei[EDGES + e]; }
  else           { s = e - EDGES; t = s; }             // self loops
}

__device__ __forceinline__ float leaky(float v) { return v > 0.f ? v : SLOPE * v; }

// ---------------- pass 1: segment max (float atomic-max via int punning) -------
__global__ void k_edge_max(const long long* __restrict__ ei, const float* __restrict__ aS,
                           const float* __restrict__ aD, float* __restrict__ mM) {
  int id = blockIdx.x * blockDim.x + threadIdx.x;      // ETOT*H threads
  if (id >= ETOT * HEADS) return;
  int e = id >> 3, h = id & 7;
  int s, t; edge_nodes(ei, e, s, t);
  float l = leaky(aS[s * 8 + h] + aD[t * 8 + h]);
  if (l >= 0.f) atomicMax((int*)(mM + t * 8 + h), __float_as_int(l));
  else          atomicMin((unsigned int*)(mM + t * 8 + h), __float_as_uint(l));
}

// ---------------- pass 2: segment sum of exp(l - max) --------------------------
__global__ void k_edge_sum(const long long* __restrict__ ei, const float* __restrict__ aS,
                           const float* __restrict__ aD, const float* __restrict__ mM,
                           float* __restrict__ den) {
  int id = blockIdx.x * blockDim.x + threadIdx.x;
  if (id >= ETOT * HEADS) return;
  int e = id >> 3, h = id & 7;
  int s, t; edge_nodes(ei, e, s, t);
  float l = leaky(aS[s * 8 + h] + aD[t * 8 + h]);
  atomicAdd(den + t * 8 + h, expf(l - mM[t * 8 + h]));
}

// ---------------- pass 3: weighted aggregation (128 lanes = one edge row) ------
__global__ void k_edge_aggr(const long long* __restrict__ ei, const float* __restrict__ aS,
                            const float* __restrict__ aD, const float* __restrict__ mM,
                            const float* __restrict__ den, const float* __restrict__ g,
                            float* __restrict__ acc) {
  int id = blockIdx.x * blockDim.x + threadIdx.x;      // ETOT*128 threads
  if (id >= ETOT * HD) return;
  int e = id >> 7, c = id & 127, h = c >> 4;
  int s, t; edge_nodes(ei, e, s, t);
  float l = leaky(aS[s * 8 + h] + aD[t * 8 + h]);
  float alpha = expf(l - mM[t * 8 + h]) / (den[t * 8 + h] + EPSD);
  atomicAdd(acc + t * HD + c, g[s * HD + c] * alpha);
}

// ---------------- regression + classifier heads --------------------------------
__global__ void k_heads(const float* __restrict__ hbuf,
                        const float* __restrict__ Wy1, const float* __restrict__ by1,
                        const float* __restrict__ Wy0, const float* __restrict__ by0,
                        const float* __restrict__ Wc1, const float* __restrict__ bc1,
                        const float* __restrict__ Wc2, const float* __restrict__ bc2,
                        float* __restrict__ out) {
  int n = blockIdx.x * blockDim.x + threadIdx.x;
  if (n >= NODES) return;
  const float* h = hbuf + n * HD;
  float y1 = by1[0], y0 = by0[0];
  float cacc[DIM];
#pragma unroll
  for (int j = 0; j < DIM; ++j) cacc[j] = bc1[j];
  for (int k = 0; k < HD; ++k) {
    float hv = h[k];
    y1 += hv * Wy1[k];
    y0 += hv * Wy0[k];
#pragma unroll
    for (int j = 0; j < DIM; ++j) cacc[j] += hv * Wc1[k * DIM + j];
  }
  float z = bc2[0];
#pragma unroll
  for (int j = 0; j < DIM; ++j) z += fmaxf(cacc[j], 0.f) * Wc2[j];
  out[n] = y1 - y0;                                  // causal effect
  out[NODES + n] = 1.f / (1.f + expf(-z));           // propensity
}

extern "C" void kernel_launch(void* const* d_in, const int* in_sizes, int n_in,
                              void* d_out, int out_size, void* d_ws, size_t ws_size,
                              hipStream_t stream) {
  (void)in_sizes; (void)n_in; (void)out_size; (void)ws_size;
  const float*     x     = (const float*)d_in[0];
  const long long* ei    = (const long long*)d_in[1];
  const float*     W_emb = (const float*)d_in[2];
  const float*     b_emb = (const float*)d_in[3];
  const float*     W0    = (const float*)d_in[4];
  const float*     as0   = (const float*)d_in[5];
  const float*     ad0   = (const float*)d_in[6];
  const float*     b0    = (const float*)d_in[7];
  const float*     W1    = (const float*)d_in[8];
  const float*     as1   = (const float*)d_in[9];
  const float*     ad1   = (const float*)d_in[10];
  const float*     b1    = (const float*)d_in[11];
  const float*     Wy1   = (const float*)d_in[12];
  const float*     by1   = (const float*)d_in[13];
  const float*     Wy0   = (const float*)d_in[14];
  const float*     by0   = (const float*)d_in[15];
  const float*     Wc1   = (const float*)d_in[16];
  const float*     bc1   = (const float*)d_in[17];
  const float*     Wc2   = (const float*)d_in[18];
  const float*     bc2   = (const float*)d_in[19];
  float* outp = (float*)d_out;

  char* p = (char*)d_ws;
  auto wsalloc = [&](size_t bytes) { char* r = p; p += (bytes + 255) & ~(size_t)255; return r; };
  float* h0  = (float*)wsalloc((size_t)NODES * DIM * 4);    // [N,16]  embedding
  float* g   = (float*)wsalloc((size_t)NODES * HD * 4);     // [N,128] per-layer projection
  float* acc = (float*)wsalloc((size_t)NODES * HD * 4);     // [N,128] layer output / next A
  float* aS  = (float*)wsalloc((size_t)NODES * HEADS * 4);
  float* aD  = (float*)wsalloc((size_t)NODES * HEADS * 4);
  float* mM  = (float*)wsalloc((size_t)NODES * HEADS * 4);
  float* den = (float*)wsalloc((size_t)NODES * HEADS * 4);
  unsigned short* Wembb = (unsigned short*)wsalloc(2048 * 2);    // K=128, NTN=1
  unsigned short* W0b   = (unsigned short*)wsalloc(4096 * 2);    // K=16 (padded), NTN=8
  unsigned short* W1b   = (unsigned short*)wsalloc(16384 * 2);   // K=128, NTN=8

  // pack weights into WMMA B-fragment order
  k_pack_b<128, 1><<<2048 / 256, 256, 0, stream>>>(W_emb, Wembb);
  k_pack_b<16,  8><<<4096 / 256, 256, 0, stream>>>(W0, W0b);
  k_pack_b<128, 8><<<16384 / 256, 256, 0, stream>>>(W1, W1b);

  const int MT = NODES / 16;                                 // 6250 exact row tiles
  const int GBLK = (MT + 7) / 8;                             // 8 waves (256 thr) per block

  // embedding: h0 = x @ W_emb + b_emb
  k_gemm<128, 1><<<GBLK, 256, 0, stream>>>(x, Wembb, h0, b_emb, MT);

  auto gat_layer = [&](const float* hin, const unsigned short* Wb, bool k16,
                       const float* attS, const float* attD, const float* bias) {
    if (k16) k_gemm<16,  8><<<GBLK, 256, 0, stream>>>(hin, Wb, g, nullptr, MT);
    else     k_gemm<128, 8><<<GBLK, 256, 0, stream>>>(hin, Wb, g, nullptr, MT);
    k_att_prep<<<(NODES * HEADS) / 256, 256, 0, stream>>>(g, attS, attD, bias,
                                                          aS, aD, mM, den, acc);
    k_edge_max <<<(ETOT * HEADS) / 256, 256, 0, stream>>>(ei, aS, aD, mM);
    k_edge_sum <<<(ETOT * HEADS) / 256, 256, 0, stream>>>(ei, aS, aD, mM, den);
    k_edge_aggr<<<(ETOT * HD)    / 256, 256, 0, stream>>>(ei, aS, aD, mM, den, g, acc);
  };

  gat_layer(h0,  W0b, true,  as0, ad0, b0);   // layer 0: K=16 (zero-padded to 32)
  gat_layer(acc, W1b, false, as1, ad1, b1);   // layer 1: K=128 (acc consumed before rewrite)

  k_heads<<<(NODES + 255) / 256, 256, 0, stream>>>(acc, Wy1, by1, Wy0, by0,
                                                   Wc1, bc1, Wc2, bc2, outp);
}
